// Model_84894323573132
// MI455X (gfx1250) — matrix operand order repaired and checked by reading, not compiled, not run on hardware
//
#include <hip/hip_runtime.h>
#include <math.h>

typedef __attribute__((ext_vector_type(16))) _Float16 v16h;
typedef __attribute__((ext_vector_type(8)))  float    v8f;

#define FEAT 512

// ---------------------------------------------------------------------------
// Scatter-mean (per edge type): sum[dst] += x[src], cnt[dst] += 1
// ---------------------------------------------------------------------------
__global__ void k_scatter_feat(const float* __restrict__ X,
                               const int* __restrict__ src,
                               const int* __restrict__ dst,
                               float* __restrict__ sum, int E)
{
    long long tid = (long long)blockIdx.x * blockDim.x + threadIdx.x;
    int e = (int)(tid >> 7);            // 128 threads per edge
    if (e >= E) return;
    int f = ((int)tid & 127) << 2;      // 4 floats per thread
    int s = src[e], d = dst[e];
    float4 v = *(const float4*)(X + (size_t)s * FEAT + f);
    float* o = sum + (size_t)d * FEAT + f;
    unsafeAtomicAdd(o + 0, v.x);
    unsafeAtomicAdd(o + 1, v.y);
    unsafeAtomicAdd(o + 2, v.z);
    unsafeAtomicAdd(o + 3, v.w);
}

__global__ void k_scatter_cnt(const int* __restrict__ dst,
                              float* __restrict__ cnt, int E)
{
    int e = blockIdx.x * blockDim.x + threadIdx.x;
    if (e >= E) return;
    unsafeAtomicAdd(cnt + dst[e], 1.0f);
}

__global__ void k_mean_div(float* __restrict__ sum, const float* __restrict__ cnt, int M)
{
    size_t i = (size_t)blockIdx.x * blockDim.x + threadIdx.x;
    if (i >= (size_t)M * FEAT) return;
    float c = cnt[i >> 9];
    sum[i] *= (c < 1.0f) ? 1.0f : (1.0f / c);
}

// ---------------------------------------------------------------------------
// Row inverse L2 norm: inv[row] = 1 / max(||X[row]||, 1e-12), one wave per row
// ---------------------------------------------------------------------------
__global__ void k_row_invnorm(const float* __restrict__ X, float* __restrict__ inv, int M)
{
    int row = (blockIdx.x << 3) + (threadIdx.x >> 5);   // 8 waves per 256-block
    int L   = threadIdx.x & 31;
    if (row >= M) return;
    const float* r = X + (size_t)row * FEAT;
    float s = 0.f;
#pragma unroll
    for (int i = 0; i < 4; ++i) {
        float4 v = *(const float4*)(r + ((i * 32 + L) << 2));
        s += v.x * v.x + v.y * v.y + v.z * v.z + v.w * v.w;
    }
#pragma unroll
    for (int off = 16; off > 0; off >>= 1) s += __shfl_xor(s, off, 32);
    if (L == 0) {
        float n = sqrtf(s);
        inv[row] = 1.0f / (n < 1e-12f ? 1e-12f : n);
    }
}

// ---------------------------------------------------------------------------
// Weight prep: pack f16 B matrices stored [N=512, K] row-major
// ---------------------------------------------------------------------------
__global__ void k_build_Benc_p(const float* __restrict__ Wr_pp, const float* __restrict__ Wr_dp,
                               const float* __restrict__ Wl_pp, const float* __restrict__ Wl_dp,
                               _Float16* __restrict__ B)
{
    int i = blockIdx.x * blockDim.x + threadIdx.x;
    if (i >= 512 * 1536) return;
    int n = i / 1536, k = i - n * 1536;
    float v;
    if (k < 512)        v = Wr_pp[(n << 9) + k];
    else if (k < 1024)  v = Wr_dp[(n << 9) + k - 512];
    else                v = Wl_pp[(n << 9) + k - 1024] + Wl_dp[(n << 9) + k - 1024];
    B[i] = (_Float16)v;
}

__global__ void k_build_Benc_d(const float* __restrict__ Wr_pd, const float* __restrict__ Wl_pd,
                               _Float16* __restrict__ B)
{
    int i = blockIdx.x * blockDim.x + threadIdx.x;
    if (i >= 512 * 1024) return;
    int n = i >> 10, k = i & 1023;
    float v = (k < 512) ? Wr_pd[(n << 9) + k] : Wl_pd[(n << 9) + k - 512];
    B[i] = (_Float16)v;
}

__global__ void k_f32_to_f16(const float* __restrict__ s, _Float16* __restrict__ d, int n)
{
    int i = blockIdx.x * blockDim.x + threadIdx.x;
    if (i < n) d[i] = (_Float16)s[i];
}

__global__ void k_bias_sum(const float* __restrict__ a, const float* __restrict__ b,
                           float* __restrict__ o)
{
    int i = blockIdx.x * blockDim.x + threadIdx.x;
    if (i < 512) o[i] = a[i] + b[i];
}

// ---------------------------------------------------------------------------
// WMMA GEMM:  C[M,512] = concatK(A0,A1,A2 each [M,512] f32, scaled per-row) @ B + bias
// B is f16, stored [N=512 rows, K=nsrc*512 cols] row-major (i.e. B_mem[n][k] = Blog[k][n]).
// Block: 256 thr = 8 waves; block tile = 16 rows x 512 cols; wave tile = 16x64.
// ---------------------------------------------------------------------------
__global__ __launch_bounds__(256) void k_gemm(
    const float* __restrict__ A0, const float* __restrict__ A1, const float* __restrict__ A2,
    const float* __restrict__ I0, const float* __restrict__ I1, const float* __restrict__ I2,
    const _Float16* __restrict__ B, const float* __restrict__ bias,
    float* __restrict__ C, int M, int nsrc)
{
    const int L  = threadIdx.x & 31;
    const int w  = threadIdx.x >> 5;    // wave 0..7
    const int hi = L >> 4;              // 0 or 1
    const int lm = L & 15;
    const int m0 = blockIdx.x << 4;
    int am = m0 + lm; if (am >= M) am = M - 1;   // clamped A row for ragged tail
    const int nb  = w << 6;                      // wave's N base (0..448)
    const int ldb = nsrc << 9;                   // K total

    v8f acc[4];
#pragma unroll
    for (int t = 0; t < 4; ++t) acc[t] = (v8f){0.f,0.f,0.f,0.f,0.f,0.f,0.f,0.f};

    for (int ks = 0; ks < nsrc; ++ks) {
        const float* As = (ks == 0) ? A0 : (ks == 1) ? A1 : A2;
        const float* Ip = (ks == 0) ? I0 : (ks == 1) ? I1 : I2;
        const float scale = Ip ? Ip[am] : 1.0f;
        const float* arow = As + (size_t)am * FEAT;
#pragma unroll 4
        for (int k = 0; k < FEAT; k += 32) {
            // A fragment (contiguous K order, matching B): lane reads 16
            // contiguous values at k + hi*16 (here f32 -> cvt f16, fused row scale)
            const float* ap = arow + k + (hi << 4);
            float4 p0 = *(const float4*)(ap);
            float4 p1 = *(const float4*)(ap + 4);
            float4 p2 = *(const float4*)(ap + 8);
            float4 p3 = *(const float4*)(ap + 12);
            v16h a;
            a[0]  = (_Float16)(p0.x * scale); a[1]  = (_Float16)(p0.y * scale);
            a[2]  = (_Float16)(p0.z * scale); a[3]  = (_Float16)(p0.w * scale);
            a[4]  = (_Float16)(p1.x * scale); a[5]  = (_Float16)(p1.y * scale);
            a[6]  = (_Float16)(p1.z * scale); a[7]  = (_Float16)(p1.w * scale);
            a[8]  = (_Float16)(p2.x * scale); a[9]  = (_Float16)(p2.y * scale);
            a[10] = (_Float16)(p2.z * scale); a[11] = (_Float16)(p2.w * scale);
            a[12] = (_Float16)(p3.x * scale); a[13] = (_Float16)(p3.y * scale);
            a[14] = (_Float16)(p3.z * scale); a[15] = (_Float16)(p3.w * scale);

            const int kg = (ks << 9) + k + (hi << 4);   // B K-offset for this lane
#pragma unroll
            for (int t = 0; t < 4; ++t) {
                const int n = nb + (t << 4) + lm;
                // B fragment (contiguous K order, matching A): lane = 16 contiguous halves of row n
                v16h b = *(const v16h*)(B + (size_t)n * ldb + kg);
                acc[t] = __builtin_amdgcn_wmma_f32_16x16x32_f16(
                    false, a, false, b, (short)0, acc[t], false, false);
            }
        }
    }

    // D layout: lane L, vgpr r -> row m0 + hi*8 + r, col nb + t*16 + (L&15)
#pragma unroll
    for (int t = 0; t < 4; ++t) {
        const int n = nb + (t << 4) + lm;
        const float bv = bias[n];
#pragma unroll
        for (int r = 0; r < 8; ++r) {
            const int row = m0 + (hi << 3) + r;
            if (row < M) C[(size_t)row * FEAT + n] = acc[t][r] + bv;
        }
    }
}

// ---------------------------------------------------------------------------
extern "C" void kernel_launch(void* const* d_in, const int* in_sizes, int n_in,
                              void* d_out, int out_size, void* d_ws, size_t ws_size,
                              hipStream_t stream)
{
    const float* x_paper   = (const float*)d_in[0];
    const float* x_dataset = (const float*)d_in[1];
    const int*   ei_pp     = (const int*)d_in[2];
    const int*   ei_pd     = (const int*)d_in[3];
    const int*   ei_dp     = (const int*)d_in[4];
    const float* Wl_pp = (const float*)d_in[5];
    const float* Wr_pp = (const float*)d_in[6];
    const float* b_pp  = (const float*)d_in[7];
    const float* Wl_pd = (const float*)d_in[8];
    const float* Wr_pd = (const float*)d_in[9];
    const float* b_pd  = (const float*)d_in[10];
    const float* Wl_dp = (const float*)d_in[11];
    const float* Wr_dp = (const float*)d_in[12];
    const float* b_dp  = (const float*)d_in[13];
    const float* Wdec_p = (const float*)d_in[14];
    const float* bdec_p = (const float*)d_in[15];
    const float* Wdec_d = (const float*)d_in[16];
    const float* bdec_d = (const float*)d_in[17];
    (void)n_in; (void)out_size; (void)ws_size;

    const int NP  = in_sizes[0] / FEAT;   // 50000 papers
    const int ND  = in_sizes[1] / FEAT;   // 5000 datasets
    const int Epp = in_sizes[2] / 2;
    const int Epd = in_sizes[3] / 2;
    const int Edp = in_sizes[4] / 2;

    // ---- workspace layout (256B aligned slices) ----
    char* ws = (char*)d_ws;
    size_t off = 0;
    auto alloc = [&](size_t bytes) -> char* {
        off = (off + 255) & ~(size_t)255;
        char* p = ws + off; off += bytes; return p;
    };
    float* meanPP = (float*)alloc((size_t)NP * FEAT * 4);
    float* meanDP = (float*)alloc((size_t)NP * FEAT * 4);
    float* meanPD = (float*)alloc((size_t)ND * FEAT * 4);
    float* cntPP  = (float*)alloc((size_t)NP * 4);
    float* cntDP  = (float*)alloc((size_t)NP * 4);
    float* cntPD  = (float*)alloc((size_t)ND * 4);
    size_t zero_span = off;                       // sums+counts zeroed each call
    float* recP   = (float*)alloc((size_t)NP * FEAT * 4);
    float* recD   = (float*)alloc((size_t)ND * FEAT * 4);
    float* invRecP = (float*)alloc((size_t)NP * 4);
    float* invXP   = (float*)alloc((size_t)NP * 4);
    float* invRecD = (float*)alloc((size_t)ND * 4);
    float* invXD   = (float*)alloc((size_t)ND * 4);
    _Float16* BencP = (_Float16*)alloc((size_t)512 * 1536 * 2);
    _Float16* BencD = (_Float16*)alloc((size_t)512 * 1024 * 2);
    _Float16* BdecP = (_Float16*)alloc((size_t)512 * 1024 * 2);
    _Float16* BdecD = (_Float16*)alloc((size_t)512 * 1024 * 2);
    float* bEncP = (float*)alloc(512 * 4);

    float* out_paper   = (float*)d_out;
    float* out_dataset = (float*)d_out + (size_t)NP * FEAT;

    // ---- 1. zero accumulators (capture-safe memset node) ----
    hipMemsetAsync(d_ws, 0, zero_span, stream);

    // ---- 2. scatter-mean per edge type ----
    auto gfeat = [](int E) { return dim3((unsigned)(((long long)E * 128 + 255) / 256)); };
    auto gcnt  = [](int E) { return dim3((unsigned)((E + 255) / 256)); };
    k_scatter_feat<<<gfeat(Epp), 256, 0, stream>>>(x_paper,   ei_pp, ei_pp + Epp, meanPP, Epp);
    k_scatter_feat<<<gfeat(Edp), 256, 0, stream>>>(x_dataset, ei_dp, ei_dp + Edp, meanDP, Edp);
    k_scatter_feat<<<gfeat(Epd), 256, 0, stream>>>(x_paper,   ei_pd, ei_pd + Epd, meanPD, Epd);
    k_scatter_cnt<<<gcnt(Epp), 256, 0, stream>>>(ei_pp + Epp, cntPP, Epp);
    k_scatter_cnt<<<gcnt(Edp), 256, 0, stream>>>(ei_dp + Edp, cntDP, Edp);
    k_scatter_cnt<<<gcnt(Epd), 256, 0, stream>>>(ei_pd + Epd, cntPD, Epd);

    auto gelem = [](long long n) { return dim3((unsigned)((n + 255) / 256)); };
    k_mean_div<<<gelem((long long)NP * FEAT), 256, 0, stream>>>(meanPP, cntPP, NP);
    k_mean_div<<<gelem((long long)NP * FEAT), 256, 0, stream>>>(meanDP, cntDP, NP);
    k_mean_div<<<gelem((long long)ND * FEAT), 256, 0, stream>>>(meanPD, cntPD, ND);

    // ---- 3. weight / bias prep (independent of scatter) ----
    k_build_Benc_p<<<gelem(512 * 1536), 256, 0, stream>>>(Wr_pp, Wr_dp, Wl_pp, Wl_dp, BencP);
    k_build_Benc_d<<<gelem(512 * 1024), 256, 0, stream>>>(Wr_pd, Wl_pd, BencD);
    k_f32_to_f16<<<gelem(512 * 1024), 256, 0, stream>>>(Wdec_p, BdecP, 512 * 1024);
    k_f32_to_f16<<<gelem(512 * 1024), 256, 0, stream>>>(Wdec_d, BdecD, 512 * 1024);
    k_bias_sum<<<dim3(2), 256, 0, stream>>>(b_pp, b_dp, bEncP);

    // raw-feature norms (independent)
    k_row_invnorm<<<dim3((NP + 7) / 8), 256, 0, stream>>>(x_paper,   invXP, NP);
    k_row_invnorm<<<dim3((ND + 7) / 8), 256, 0, stream>>>(x_dataset, invXD, ND);

    // ---- 4. encoder GEMMs ----
    // rec_paper = [meanPP|meanDP|x_paper] @ [Wr_pp;Wr_dp;Wl_pp+Wl_dp]^T + (b_pp+b_dp)
    k_gemm<<<dim3((NP + 15) / 16), 256, 0, stream>>>(
        meanPP, meanDP, x_paper, nullptr, nullptr, nullptr,
        BencP, bEncP, recP, NP, 3);
    // rec_dataset = [meanPD|x_dataset] @ [Wr_pd;Wl_pd]^T + b_pd
    k_gemm<<<dim3((ND + 15) / 16), 256, 0, stream>>>(
        meanPD, x_dataset, nullptr, nullptr, nullptr, nullptr,
        BencD, b_pd, recD, ND, 2);

    // ---- 5. rec norms ----
    k_row_invnorm<<<dim3((NP + 7) / 8), 256, 0, stream>>>(recP, invRecP, NP);
    k_row_invnorm<<<dim3((ND + 7) / 8), 256, 0, stream>>>(recD, invRecD, ND);

    // ---- 6. decoder GEMMs (L2 norm folded into A-load via per-row scales) ----
    k_gemm<<<dim3((NP + 15) / 16), 256, 0, stream>>>(
        recP, x_paper, nullptr, invRecP, invXP, nullptr,
        BdecP, bdec_p, out_paper, NP, 2);
    k_gemm<<<dim3((ND + 15) / 16), 256, 0, stream>>>(
        recD, x_dataset, nullptr, invRecD, invXD, nullptr,
        BdecD, bdec_d, out_dataset, ND, 2);
}
